// GRUBackbone_72825465471358
// MI455X (gfx1250) — compile-verified, matrix operand order checked by reading, not run on hardware
//
#include <hip/hip_runtime.h>
#include <stdint.h>

// ---------------------------------------------------------------------------
// 2-layer GRU backbone on gfx1250 (MI455X), bf16 WMMA with fp32 accumulate.
// All GEMM operands kept in bf16 in memory so fragments are straight b128
// loads (no per-use conversion VALU). fp32 carries kept only for the z*h term.
// ---------------------------------------------------------------------------

typedef __bf16 bf16_t;
typedef __attribute__((ext_vector_type(16))) __bf16 v16bf;
typedef __attribute__((ext_vector_type(8)))  float  v8f;

#define GRU_S  512
#define GRU_B  256
#define GRU_IN 256
#define GRU_H  512

static __device__ __forceinline__ unsigned short f2bf_bits(float f) {
  unsigned u; __builtin_memcpy(&u, &f, 4);
  u = (u + 0x7FFFu + ((u >> 16) & 1u)) >> 16;     // round-to-nearest-even
  return (unsigned short)u;
}

// grid-stride fp32 -> bf16 bit conversion (weights / initial hidden / x[0])
__global__ void f32_to_bf16_kernel(const float* __restrict__ src,
                                   unsigned short* __restrict__ dst, int n) {
  const int stride = gridDim.x * blockDim.x;
  for (int i = blockIdx.x * blockDim.x + threadIdx.x; i < n; i += stride)
    dst[i] = f2bf_bits(src[i]);
}

// Fragment load: bf16 memory already matches the ISA 16-bit A/B layout for
// this lane; element 0..7 at p[0..7], 8..15 at p[16..23]. Two b128 loads.
static __device__ __forceinline__ v16bf load_frag(const unsigned short* p) {
  v16bf v;
  __builtin_memcpy(&v, p, 16);
  __builtin_memcpy(reinterpret_cast<char*>(&v) + 16, p + 16, 16);
  return v;
}

static __device__ __forceinline__ v8f wmma_bf16(v16bf a, v16bf b, v8f c) {
  return __builtin_amdgcn_wmma_f32_16x16x32_bf16(
      /*neg_a=*/false, a, /*neg_b=*/false, b,
      /*c_mod=*/(short)0, c, /*reuse_a=*/false, /*reuse_b=*/false);
}

// One GRU cell step for one layer (fused gates).
//   gi = a @ wih^T + bi ; gh = h @ whh^T + bh
//   r = sig(gi_r+gh_r); z = sig(gi_z+gh_z); n = tanh(gi_n + r*gh_n)
//   h' = (1-z)*n + z*h ;  carry = h' * (1-mask)
// 512 waves: 16 batch-tiles x 32 H-col-tiles; one 16x16 tile per wave with
// six f32 accumulators. Optionally converts next step's x_t to bf16 (fused).
template <int KI>
__global__ __launch_bounds__(256) void gru_step_kernel(
    const unsigned short* __restrict__ a_bf,     // (256, KI) activations bf16
    const float*          __restrict__ hcur,     // (256, 512) carry fp32
    const unsigned short* __restrict__ hcur_bf,  // (256, 512) carry bf16
    const unsigned short* __restrict__ wih,      // (1536, KI) bf16
    const unsigned short* __restrict__ whh,      // (1536, 512) bf16
    const float* __restrict__ bi,                // (1536)
    const float* __restrict__ bh,                // (1536)
    const float* __restrict__ mask_s,            // (256)
    float*          __restrict__ hpre_f32,       // nullable: pre-mask out fp32
    unsigned short* __restrict__ hpre_bf,        // nullable: pre-mask out bf16
    float*          __restrict__ hcarry_f32,     // masked carry fp32
    unsigned short* __restrict__ hcarry_bf,      // masked carry bf16
    const float*    __restrict__ xnext,          // nullable: x_{s+1} fp32
    unsigned short* __restrict__ xnext_bf)       // bf16 dest for x_{s+1}
{
  const int H = GRU_H;

  // ---- fused conversion of next timestep's input (hidden under this GEMM)
  if (xnext) {
    const int stride = gridDim.x * blockDim.x;
    for (int i = blockIdx.x * blockDim.x + threadIdx.x; i < GRU_B * GRU_IN;
         i += stride)
      xnext_bf[i] = f2bf_bits(xnext[i]);
  }

  const int tid  = blockIdx.x * blockDim.x + threadIdx.x;
  const int wave = tid >> 5;
  const int lane = threadIdx.x & 31;
  const int half = lane >> 4;
  const int n16  = lane & 15;
  const int row_tile = wave & 15;          // B/16 = 16
  const int col_tile = wave >> 4;          // H/16 = 32
  const int row_base = row_tile << 4;
  const int colg = (col_tile << 4) + n16;  // H column for this lane
  const int koff = 8 * half;               // lane K sub-offset

  v8f acc[6];
  const v8f vzero = {0.f, 0.f, 0.f, 0.f, 0.f, 0.f, 0.f, 0.f};
#pragma unroll
  for (int i = 0; i < 6; ++i) acc[i] = vzero;

  // ---- phase 0: gi = a @ wih^T   (K = KI)
  {
    const unsigned short* arow = a_bf + (size_t)(row_base + n16) * KI + koff;
    const unsigned short* w0r = wih + (size_t)(0 * H + colg) * KI + koff;
    const unsigned short* w1r = wih + (size_t)(1 * H + colg) * KI + koff;
    const unsigned short* w2r = wih + (size_t)(2 * H + colg) * KI + koff;
#pragma unroll 4
    for (int k0 = 0; k0 < KI; k0 += 32) {
      v16bf Af = load_frag(arow + k0);
      __builtin_prefetch(w0r + k0 + 32, 0, 3);
      __builtin_prefetch(w1r + k0 + 32, 0, 3);
      __builtin_prefetch(w2r + k0 + 32, 0, 3);
      acc[0] = wmma_bf16(Af, load_frag(w0r + k0), acc[0]);
      acc[1] = wmma_bf16(Af, load_frag(w1r + k0), acc[1]);
      acc[2] = wmma_bf16(Af, load_frag(w2r + k0), acc[2]);
    }
  }
  // ---- phase 1: gh = h @ whh^T   (K = H)
  {
    const unsigned short* arow = hcur_bf + (size_t)(row_base + n16) * H + koff;
    const unsigned short* w0r = whh + (size_t)(0 * H + colg) * H + koff;
    const unsigned short* w1r = whh + (size_t)(1 * H + colg) * H + koff;
    const unsigned short* w2r = whh + (size_t)(2 * H + colg) * H + koff;
#pragma unroll 4
    for (int k0 = 0; k0 < H; k0 += 32) {
      v16bf Af = load_frag(arow + k0);
      __builtin_prefetch(w0r + k0 + 32, 0, 3);
      __builtin_prefetch(w1r + k0 + 32, 0, 3);
      __builtin_prefetch(w2r + k0 + 32, 0, 3);
      acc[3] = wmma_bf16(Af, load_frag(w0r + k0), acc[3]);
      acc[4] = wmma_bf16(Af, load_frag(w1r + k0), acc[4]);
      acc[5] = wmma_bf16(Af, load_frag(w2r + k0), acc[5]);
    }
  }

  // ---- fused gate math; C/D layout: element r -> row (r + 8*half), col colg
  const float biR = bi[colg], biZ = bi[H + colg], biN = bi[2 * H + colg];
  const float bhR = bh[colg], bhZ = bh[H + colg], bhN = bh[2 * H + colg];
#pragma unroll
  for (int r = 0; r < 8; ++r) {
    const int brow = row_base + r + 8 * half;         // batch index
    float ir = acc[0][r] + biR, iz = acc[1][r] + biZ, inn = acc[2][r] + biN;
    float hr = acc[3][r] + bhR, hz = acc[4][r] + bhZ, hn  = acc[5][r] + bhN;
    float rg = 1.f / (1.f + __expf(-(ir + hr)));
    float zg = 1.f / (1.f + __expf(-(iz + hz)));
    float ng = tanhf(inn + rg * hn);
    const size_t o = (size_t)brow * H + colg;
    float hnew = (1.f - zg) * ng + zg * hcur[o];
    float hc   = hnew * (1.f - mask_s[brow]);
    hcarry_f32[o] = hc;
    hcarry_bf[o]  = f2bf_bits(hc);
    if (hpre_f32) hpre_f32[o] = hnew;
    if (hpre_bf)  hpre_bf[o]  = f2bf_bits(hnew);
  }
}

extern "C" void kernel_launch(void* const* d_in, const int* in_sizes, int n_in,
                              void* d_out, int out_size, void* d_ws, size_t ws_size,
                              hipStream_t stream) {
  const int S = GRU_S, B = GRU_B, IN = GRU_IN, H = GRU_H;
  const float* x      = (const float*)d_in[0];
  const float* hidden = (const float*)d_in[1];
  const float* mask   = (const float*)d_in[2];
  const float* wih0   = (const float*)d_in[3];
  const float* whh0   = (const float*)d_in[4];
  const float* bih0   = (const float*)d_in[5];
  const float* bhh0   = (const float*)d_in[6];
  const float* wih1   = (const float*)d_in[7];
  const float* whh1   = (const float*)d_in[8];
  const float* bih1   = (const float*)d_in[9];
  const float* bhh1   = (const float*)d_in[10];

  float* out    = (float*)d_out;                 // (S, B, H)
  float* hfinal = out + (size_t)S * B * H;       // (L, B, H)

  // workspace carve-out (~9 MB)
  char*  ws  = (char*)d_ws;
  size_t off = 0;
  auto carve = [&](size_t bytes) -> void* {
    void* p = ws + off;
    off += (bytes + 255) & ~(size_t)255;
    return p;
  };
  unsigned short* wih0b = (unsigned short*)carve((size_t)3 * H * IN * 2);
  unsigned short* whh0b = (unsigned short*)carve((size_t)3 * H * H  * 2);
  unsigned short* wih1b = (unsigned short*)carve((size_t)3 * H * H  * 2);
  unsigned short* whh1b = (unsigned short*)carve((size_t)3 * H * H  * 2);
  float* h0c[2]  = { (float*)carve((size_t)B * H * 4), (float*)carve((size_t)B * H * 4) };
  float* h1c[2]  = { (float*)carve((size_t)B * H * 4), (float*)carve((size_t)B * H * 4) };
  unsigned short* h0cb[2] = { (unsigned short*)carve((size_t)B * H * 2),
                              (unsigned short*)carve((size_t)B * H * 2) };
  unsigned short* h1cb[2] = { (unsigned short*)carve((size_t)B * H * 2),
                              (unsigned short*)carve((size_t)B * H * 2) };
  unsigned short* h0outb = (unsigned short*)carve((size_t)B * H * 2);
  unsigned short* xbf    = (unsigned short*)carve((size_t)B * IN * 2);

  // one-time conversions (weights stay resident in 192MB L2 across all steps)
  auto conv = [&](const float* s, unsigned short* d, int n) {
    f32_to_bf16_kernel<<<256, 256, 0, stream>>>(s, d, n);
  };
  conv(wih0, wih0b, 3 * H * IN);
  conv(whh0, whh0b, 3 * H * H);
  conv(wih1, wih1b, 3 * H * H);
  conv(whh1, whh1b, 3 * H * H);
  conv(hidden,                 h0cb[0], B * H);
  conv(hidden + (size_t)B * H, h1cb[0], B * H);
  conv(x, xbf, B * IN);                                  // x_0

  hipMemcpyAsync(h0c[0], hidden,                 (size_t)B * H * 4,
                 hipMemcpyDeviceToDevice, stream);
  hipMemcpyAsync(h1c[0], hidden + (size_t)B * H, (size_t)B * H * 4,
                 hipMemcpyDeviceToDevice, stream);

  // 512 waves per step-kernel: 16 batch-tiles x 32 col-tiles
  dim3 grid(64), block(256);
  for (int s = 0; s < S; ++s) {
    const int cur = s & 1, nxt = cur ^ 1;
    // layer 0: A = x_s(bf16); pre-mask -> h0outb (bf16 only, feeds layer 1)
    gru_step_kernel<GRU_IN><<<grid, block, 0, stream>>>(
        xbf, h0c[cur], h0cb[cur], wih0b, whh0b, bih0, bhh0,
        mask + (size_t)s * B,
        /*hpre_f32=*/nullptr, /*hpre_bf=*/h0outb,
        h0c[nxt], h0cb[nxt],
        /*xnext=*/nullptr, /*xnext_bf=*/nullptr);
    // layer 1: A = h0outb; pre-mask -> d_out[s] (fp32); fuse x_{s+1} conversion
    const float* xn = (s + 1 < S) ? (x + (size_t)(s + 1) * B * IN) : nullptr;
    gru_step_kernel<GRU_H><<<grid, block, 0, stream>>>(
        h0outb, h1c[cur], h1cb[cur], wih1b, whh1b, bih1, bhh1,
        mask + (size_t)s * B,
        /*hpre_f32=*/out + (size_t)s * B * H, /*hpre_bf=*/nullptr,
        h1c[nxt], h1cb[nxt],
        /*xnext=*/xn, /*xnext_bf=*/xbf);
  }

  // final hidden state (last write went to index S&1 == 0)
  hipMemcpyAsync(hfinal,                 h0c[S & 1], (size_t)B * H * 4,
                 hipMemcpyDeviceToDevice, stream);
  hipMemcpyAsync(hfinal + (size_t)B * H, h1c[S & 1], (size_t)B * H * 4,
                 hipMemcpyDeviceToDevice, stream);
}
